// TranHGAT_49984829391163
// MI455X (gfx1250) — matrix operand order repaired and checked by reading, not compiled, not run on hardware
//
#include <hip/hip_runtime.h>
#include <cstddef>
#include <cstdint>

// ---------------- problem dims ----------------
#define ATTR_N 5
#define BB     32
#define SS     128
#define VV     30522
#define HD     768          // hidden
#define NHEAD  12
#define DHEAD  64
#define FFD    3072
#define NLAYER 4
#define NTOK   (BB * SS)    // 4096 rows

typedef __attribute__((ext_vector_type(16))) _Float16 v16h;
typedef __attribute__((ext_vector_type(8)))  float    v8f;
typedef __attribute__((ext_vector_type(4)))  int      v4i;

union FragAB { unsigned u[8]; v16h v; };

#if defined(__HIP_DEVICE_COMPILE__) && __has_builtin(__builtin_amdgcn_global_load_async_to_lds_b128)
#define HAVE_ASYNC_LDS 1
#else
#define HAVE_ASYNC_LDS 0
#endif

#if HAVE_ASYNC_LDS
// builtin expects v4i* operands (16B granule); cast via void* to drop const/type
#define GAS1(p) ((__attribute__((address_space(1))) v4i*)(void*)(p))
#define LAS3(p) ((__attribute__((address_space(3))) v4i*)(void*)(p))
#if __has_builtin(__builtin_amdgcn_s_wait_asynccnt)
#define WAIT_ASYNC0() __builtin_amdgcn_s_wait_asynccnt(0)
#else
#define WAIT_ASYNC0() asm volatile("s_wait_asynccnt 0x0" ::: "memory")
#endif
#endif

__device__ inline float leaky_f(float x) { return x >= 0.f ? x : 0.2f * x; }
__device__ inline float gelu_exact(float x) { return 0.5f * x * (1.0f + erff(x * 0.70710678118654752f)); }

__device__ inline float wave_reduce_sum(float v) {
  #pragma unroll
  for (int off = 16; off; off >>= 1) v += __shfl_xor(v, off, 32);
  return v;
}

// ---------------- WMMA GEMM: C = act(A[MxK] * W[KxN] + bias[N]) ----------------
// f16 operands, f32 accumulate. Block tile 128(M) x 128(N), K step 32.
// 8 waves: 4 (M) x 2 (N); each wave owns 32x64 = 2x4 WMMA 16x16x32 tiles.
// Double-buffered LDS fed by gfx1250 async-to-LDS B128 copies (ASYNCcnt):
//   prefetch(tile k+1) overlaps the 8 WMMAs of tile k; one barrier/iteration.
// Row strides 56/136 halfs keep every per-lane B128 destination 16B aligned.
template <int ACT>
__global__ __launch_bounds__(256) void gemm_wmma_kernel(
    const _Float16* __restrict__ A, const _Float16* __restrict__ W,
    const float* __restrict__ bias, float* __restrict__ C,
    _Float16* __restrict__ C16, int M, int N, int K)
{
  __shared__ _Float16 sA[2][128][56];   // 32 used, stride 112B (16B-aligned rows)
  __shared__ _Float16 sB[2][32][136];   // 128 used, stride 272B (16B-aligned rows)

  const int tid  = threadIdx.x;
  const int lane = tid & 31;
  const int wave = tid >> 5;
  const int wm = (wave & 3) << 5;          // wave M offset in block tile
  const int wn = (wave >> 2) << 6;         // wave N offset in block tile
  const int rowBase = blockIdx.y << 7;
  const int colBase = blockIdx.x << 7;

#if HAVE_ASYNC_LDS
  // A tile: 128 rows x 64B = 512 x 16B chunks; B tile: 32 rows x 256B = 512 chunks
  auto prefetch = [&](int kb, int buf) {
    int t = tid;                                   // chunk 0..255
    int rA = t >> 2, cA = t & 3;
    __builtin_amdgcn_global_load_async_to_lds_b128(
        GAS1(A + (size_t)(rowBase + rA) * K + kb + cA * 8), LAS3(&sA[buf][rA][cA * 8]), 0, 0);
    int rB = t >> 4, cB = t & 15;
    __builtin_amdgcn_global_load_async_to_lds_b128(
        GAS1(W + (size_t)(kb + rB) * N + colBase + cB * 8), LAS3(&sB[buf][rB][cB * 8]), 0, 0);
    t = tid + 256;                                 // chunk 256..511
    rA = t >> 2; cA = t & 3;
    __builtin_amdgcn_global_load_async_to_lds_b128(
        GAS1(A + (size_t)(rowBase + rA) * K + kb + cA * 8), LAS3(&sA[buf][rA][cA * 8]), 0, 0);
    rB = t >> 4; cB = t & 15;
    __builtin_amdgcn_global_load_async_to_lds_b128(
        GAS1(W + (size_t)(kb + rB) * N + colBase + cB * 8), LAS3(&sB[buf][rB][cB * 8]), 0, 0);
  };
#endif

  v8f acc[2][4];
  #pragma unroll
  for (int i = 0; i < 2; ++i)
    #pragma unroll
    for (int j = 0; j < 4; ++j)
      acc[i][j] = (v8f){0,0,0,0,0,0,0,0};

  const int nk = K >> 5;
#if HAVE_ASYNC_LDS
  prefetch(0, 0);
#endif
  for (int ik = 0; ik < nk; ++ik) {
    const int p = ik & 1;
#if HAVE_ASYNC_LDS
    WAIT_ASYNC0();                 // own share of tile ik landed
    __syncthreads();               // everyone's tile ik landed; prev reads of buf 1-p done
    if (ik + 1 < nk) prefetch((ik + 1) << 5, 1 - p);
#else
    const int kb = ik << 5;
    #pragma unroll
    for (int i = 0; i < 8; ++i) {                  // A: 2048 dwords
      int idx = (i << 8) + tid;
      int r = idx >> 4, c = idx & 15;
      *(unsigned*)&sA[p][r][c << 1] =
          ((const unsigned*)(A + (size_t)(rowBase + r) * K + kb))[c];
    }
    #pragma unroll
    for (int i = 0; i < 8; ++i) {                  // B: 2048 dwords
      int idx = (i << 8) + tid;
      int r = idx >> 6, c = idx & 63;
      *(unsigned*)&sB[p][r][c << 1] =
          ((const unsigned*)(W + (size_t)(kb + r) * N + colBase))[c];
    }
    __syncthreads();
#endif

    FragAB af[2], bf[4];
    #pragma unroll
    for (int sm = 0; sm < 2; ++sm) {
      const int ar = wm + (sm << 4) + (lane & 15);
      #pragma unroll
      for (int j = 0; j < 8; ++j) {
        const int kk = ((j < 4) ? 0 : 16) + ((lane < 16) ? 0 : 8) + ((j & 3) << 1);
        af[sm].u[j] = *(const unsigned*)&sA[p][ar][kk];
      }
    }
    #pragma unroll
    for (int sn = 0; sn < 4; ++sn) {
      const int nb = wn + (sn << 4);
      #pragma unroll
      for (int j = 0; j < 8; ++j) {
        const int kk = ((j < 4) ? 0 : 16) + ((lane < 16) ? 0 : 8) + ((j & 3) << 1);
        bf[sn].u[j] = (unsigned)*(const unsigned short*)&sB[p][kk][nb + (lane & 15)]
                    | ((unsigned)*(const unsigned short*)&sB[p][kk + 1][nb + (lane & 15)] << 16);
      }
    }
    #pragma unroll
    for (int sm = 0; sm < 2; ++sm)
      #pragma unroll
      for (int sn = 0; sn < 4; ++sn)
        acc[sm][sn] = __builtin_amdgcn_wmma_f32_16x16x32_f16(
            false, af[sm].v, false, bf[sn].v, (short)0, acc[sm][sn], false, false);
  }

  // store: C tile layout: VGPR r -> M = r + (lane>=16 ? 8 : 0); N = lane & 15
  const int mOff = (lane < 16) ? 0 : 8;
  const int nOff = lane & 15;
  #pragma unroll
  for (int sm = 0; sm < 2; ++sm) {
    #pragma unroll
    for (int sn = 0; sn < 4; ++sn) {
      const int n = colBase + wn + (sn << 4) + nOff;
      const float bv = bias[n];
      const int mBase = rowBase + wm + (sm << 4) + mOff;
      #pragma unroll
      for (int r = 0; r < 8; ++r) {
        float v = acc[sm][sn][r] + bv;
        if (ACT == 1) v = gelu_exact(v);
        if (C)   C[(size_t)(mBase + r) * N + n] = v;
        if (C16) C16[(size_t)(mBase + r) * N + n] = (_Float16)v;
      }
    }
  }
}

// ---------------- small kernels ----------------
__global__ void zero_kernel(float* __restrict__ p, int n) {
  int i = blockIdx.x * blockDim.x + threadIdx.x;
  if (i < n) p[i] = 0.f;
}

__global__ void cvt_f16_kernel(const float* __restrict__ src, _Float16* __restrict__ dst, int n) {
  int i = blockIdx.x * blockDim.x + threadIdx.x;
  if (i < n) dst[i] = (_Float16)src[i];
}

__global__ void gather_we_kernel(const int* __restrict__ xs, const float* __restrict__ wordEmb,
                                 float* __restrict__ we, _Float16* __restrict__ we16, int a)
{
  int idx = blockIdx.x * blockDim.x + threadIdx.x;
  if (idx >= NTOK * HD) return;
  int row = idx / HD;
  int c   = idx % HD;
  int b = row >> 7, s = row & 127;
  int w = xs[(b * ATTR_N + a) * SS + s];
  float v = wordEmb[(size_t)w * HD + c];
  we[(size_t)idx] = v;
  we16[(size_t)idx] = (_Float16)v;
}

// wave per row: e[row] = leaky(dot(X[row,:768], avec))
__global__ __launch_bounds__(256) void rowdot_leaky_kernel(
    const float* __restrict__ X, const float* __restrict__ avec, float* __restrict__ e)
{
  int lane = threadIdx.x & 31;
  int row  = blockIdx.x * 8 + (threadIdx.x >> 5);
  float s = 0.f;
  #pragma unroll
  for (int i = 0; i < 24; ++i) {
    int c = lane + (i << 5);
    s += X[(size_t)row * HD + c] * avec[c];
  }
  s = wave_reduce_sum(s);
  if (lane == 0) e[row] = leaky_f(s);
}

// block per b: softmax over S of e, attr_emb = relu(sum_s att*hh)
__global__ __launch_bounds__(128) void attr_pool_kernel(
    const float* __restrict__ e, const float* __restrict__ hh,
    float* __restrict__ attrOuts, int a)
{
  __shared__ float red[128];
  __shared__ float sAtt[128];
  int b = blockIdx.x, tid = threadIdx.x;
  float v = e[b * SS + tid];
  red[tid] = v; __syncthreads();
  for (int off = 64; off; off >>= 1) { if (tid < off) red[tid] = fmaxf(red[tid], red[tid + off]); __syncthreads(); }
  float mx = red[0]; __syncthreads();
  float p = __expf(v - mx);
  red[tid] = p; __syncthreads();
  for (int off = 64; off; off >>= 1) { if (tid < off) red[tid] += red[tid + off]; __syncthreads(); }
  float ssum = red[0]; __syncthreads();
  sAtt[tid] = p / ssum; __syncthreads();
  #pragma unroll
  for (int i = 0; i < 6; ++i) {
    int c = tid + (i << 7);
    float acc = 0.f;
    for (int s2 = 0; s2 < SS; ++s2) acc += sAtt[s2] * hh[((size_t)b * SS + s2) * HD + c];
    attrOuts[((size_t)b * ATTR_N + a) * HD + c] = fmaxf(acc, 0.f);
  }
}

// wave per (b,s): e2 = leaky(we.ca[0:768] + attr_emb.ca[768:1536]); scatter into vocab table
__global__ __launch_bounds__(256) void e2_scatter_kernel(
    const float* __restrict__ we, const float* __restrict__ attrOuts,
    const float* __restrict__ ca, const int* __restrict__ xs,
    float* __restrict__ table, int a)
{
  int lane = threadIdx.x & 31;
  int row  = blockIdx.x * 8 + (threadIdx.x >> 5);
  int b = row >> 7, s = row & 127;
  float acc = 0.f;
  #pragma unroll
  for (int i = 0; i < 24; ++i) {
    int c = lane + (i << 5);
    acc += we[(size_t)row * HD + c] * ca[c];
    acc += attrOuts[((size_t)b * ATTR_N + a) * HD + c] * ca[HD + c];
  }
  acc = wave_reduce_sum(acc);
  if (lane == 0) {
    int w = xs[(b * ATTR_N + a) * SS + s];
    table[((size_t)b * VV + w) * ATTR_N + a] = leaky_f(acc);
  }
}

// block per (b,s): we' = W[w] + sum_a' softmax_a'(table[b,w,:])*mask * attrOuts + pos + type
__global__ __launch_bounds__(256) void build_input_kernel(
    const int* __restrict__ xs, const float* __restrict__ wordEmb,
    const float* __restrict__ posEmb, const float* __restrict__ typeEmb,
    const float* __restrict__ table, const float* __restrict__ masks,
    const float* __restrict__ attrOuts, float* __restrict__ out, int a)
{
  int row = blockIdx.x;
  int b = row >> 7, s = row & 127;
  int w = xs[(b * ATTR_N + a) * SS + s];
  float t[ATTR_N], mx = -3.0e38f;
  #pragma unroll
  for (int a2 = 0; a2 < ATTR_N; ++a2) {
    t[a2] = table[((size_t)b * VV + w) * ATTR_N + a2];
    mx = fmaxf(mx, t[a2]);
  }
  float sum = 0.f;
  #pragma unroll
  for (int a2 = 0; a2 < ATTR_N; ++a2) { t[a2] = __expf(t[a2] - mx); sum += t[a2]; }
  float g[ATTR_N];
  #pragma unroll
  for (int a2 = 0; a2 < ATTR_N; ++a2)
    g[a2] = t[a2] / sum * masks[((size_t)b * ATTR_N + a2) * VV + w];
  for (int c = threadIdx.x; c < HD; c += 256) {
    float v = wordEmb[(size_t)w * HD + c] + posEmb[s * HD + c] + typeEmb[c];
    #pragma unroll
    for (int a2 = 0; a2 < ATTR_N; ++a2)
      v += g[a2] * attrOuts[((size_t)b * ATTR_N + a2) * HD + c];
    out[(size_t)row * HD + c] = v;
  }
}

// wave per row LayerNorm over 768, optional residual, optional f16 duplicate output
__global__ __launch_bounds__(256) void ln_kernel(
    float* __restrict__ out, _Float16* __restrict__ out16,
    const float* __restrict__ in, const float* __restrict__ res,
    const float* __restrict__ gam, const float* __restrict__ bet)
{
  int lane = threadIdx.x & 31;
  int row  = blockIdx.x * 8 + (threadIdx.x >> 5);
  float x[24];
  float s = 0.f;
  #pragma unroll
  for (int i = 0; i < 24; ++i) {
    int c = lane + (i << 5);
    float v = in[(size_t)row * HD + c];
    if (res) v += res[(size_t)row * HD + c];
    x[i] = v; s += v;
  }
  s = wave_reduce_sum(s);
  float mean = s * (1.f / 768.f);
  float vs = 0.f;
  #pragma unroll
  for (int i = 0; i < 24; ++i) { float d = x[i] - mean; vs += d * d; }
  vs = wave_reduce_sum(vs);
  float inv = rsqrtf(vs * (1.f / 768.f) + 1e-12f);
  #pragma unroll
  for (int i = 0; i < 24; ++i) {
    int c = lane + (i << 5);
    float v = (x[i] - mean) * inv * gam[c] + bet[c];
    out[(size_t)row * HD + c] = v;
    if (out16) out16[(size_t)row * HD + c] = (_Float16)v;
  }
}

// block per (b,head): softmax(QK^T/8)V -> f16 ctx; K/V in LDS, one thread per query row
__global__ __launch_bounds__(128) void attention_kernel(
    const float* __restrict__ Q, const float* __restrict__ K,
    const float* __restrict__ V, _Float16* __restrict__ Ctx16)
{
  __shared__ float sK[SS][DHEAD];
  __shared__ float sV[SS][DHEAD];
  const int bh = blockIdx.x;
  const int b = bh / NHEAD, hd = bh % NHEAD;
  const int tid = threadIdx.x;
  const size_t base = (size_t)b * SS * HD + (size_t)hd * DHEAD;
  #pragma unroll 4
  for (int i = 0; i < 64; ++i) {
    int idx = (i << 7) + tid;
    int j = idx >> 6, d = idx & 63;
    sK[j][d] = K[base + (size_t)j * HD + d];
    sV[j][d] = V[base + (size_t)j * HD + d];
  }
  __syncthreads();
  float q[DHEAD];
  #pragma unroll
  for (int d = 0; d < DHEAD; ++d) q[d] = Q[base + (size_t)tid * HD + d];
  const float scale = 0.125f;
  float mx = -3.0e38f;
  for (int j = 0; j < SS; ++j) {
    float s = 0.f;
    #pragma unroll
    for (int d = 0; d < DHEAD; ++d) s += q[d] * sK[j][d];
    mx = fmaxf(mx, s * scale);
  }
  float ctx[DHEAD];
  #pragma unroll
  for (int d = 0; d < DHEAD; ++d) ctx[d] = 0.f;
  float sum = 0.f;
  for (int j = 0; j < SS; ++j) {
    float s = 0.f;
    #pragma unroll
    for (int d = 0; d < DHEAD; ++d) s += q[d] * sK[j][d];
    float p = __expf(s * scale - mx);
    sum += p;
    #pragma unroll
    for (int d = 0; d < DHEAD; ++d) ctx[d] += p * sV[j][d];
  }
  float inv = 1.f / sum;
  #pragma unroll
  for (int d = 0; d < DHEAD; ++d)
    Ctx16[base + (size_t)tid * HD + d] = (_Float16)(ctx[d] * inv);
}

__global__ void copy_cls_kernel(const float* __restrict__ h, float* __restrict__ pooled, int a) {
  int b = blockIdx.x;
  for (int c = threadIdx.x; c < HD; c += 256)
    pooled[((size_t)b * ATTR_N + a) * HD + c] = h[(size_t)(b * SS) * HD + c];
}

// block per b: out-attention + entity + logits
__global__ __launch_bounds__(256) void final_kernel(
    const float* __restrict__ pooled, const float* __restrict__ outA,
    const float* __restrict__ fcW, const float* __restrict__ fcB,
    float* __restrict__ dout)
{
  __shared__ float red[256];
  __shared__ float e3s[ATTR_N];
  __shared__ float att3[ATTR_N];
  __shared__ float ent[HD];
  int b = blockIdx.x, tid = threadIdx.x;
  float acc = 0.f;
  for (int i = tid; i < ATTR_N * HD; i += 256)
    acc += pooled[(size_t)b * ATTR_N * HD + i] * outA[HD + i];
  red[tid] = acc; __syncthreads();
  for (int off = 128; off; off >>= 1) { if (tid < off) red[tid] += red[tid + off]; __syncthreads(); }
  float common = red[0]; __syncthreads();
  for (int a = 0; a < ATTR_N; ++a) {
    float a2 = 0.f;
    for (int i = tid; i < HD; i += 256)
      a2 += pooled[((size_t)b * ATTR_N + a) * HD + i] * outA[i];
    red[tid] = a2; __syncthreads();
    for (int off = 128; off; off >>= 1) { if (tid < off) red[tid] += red[tid + off]; __syncthreads(); }
    if (tid == 0) e3s[a] = leaky_f(red[0] + common);
    __syncthreads();
  }
  if (tid == 0) {
    float mx = -3.0e38f;
    for (int a = 0; a < ATTR_N; ++a) mx = fmaxf(mx, e3s[a]);
    float s = 0.f;
    for (int a = 0; a < ATTR_N; ++a) { att3[a] = __expf(e3s[a] - mx); s += att3[a]; }
    for (int a = 0; a < ATTR_N; ++a) att3[a] = att3[a] / s * (float)ATTR_N;
  }
  __syncthreads();
  for (int c = tid; c < HD; c += 256) {
    float v = 0.f;
    #pragma unroll
    for (int a = 0; a < ATTR_N; ++a)
      v += att3[a] * pooled[((size_t)b * ATTR_N + a) * HD + c];
    ent[c] = v;
    dout[(size_t)b * HD + c] = v;
  }
  __syncthreads();
  for (int k = 0; k < 2; ++k) {
    float a2 = 0.f;
    for (int c = tid; c < HD; c += 256) a2 += ent[c] * fcW[c * 2 + k];
    red[tid] = a2; __syncthreads();
    for (int off = 128; off; off >>= 1) { if (tid < off) red[tid] += red[tid + off]; __syncthreads(); }
    if (tid == 0) dout[(size_t)BB * HD + b * 2 + k] = red[0] + fcB[k];
    __syncthreads();
  }
}

// ---------------- launch ----------------
extern "C" void kernel_launch(void* const* d_in, const int* in_sizes, int n_in,
                              void* d_out, int out_size, void* d_ws, size_t ws_size,
                              hipStream_t stream)
{
  (void)in_sizes; (void)n_in; (void)out_size; (void)ws_size;
  // pytree flattening (insertion order, depth first):
  // 0 xs, 1 masks, 2 word_emb, 3 pos_emb, 4 type_emb, 5 emb_ln_g, 6 emb_ln_b,
  // 7..70 layers[l]{wq,bq,wk,bk,wv,bv,wo,bo,ln1_g,ln1_b,w1,b1,w2,b2,ln2_g,ln2_b},
  // 71..85 inits[a]{w,b,a}, 86..90 conts[a]{a}, 91 out_a, 92 fc_w, 93 fc_b
  const int*   xs      = (const int*)d_in[0];
  const float* masks   = (const float*)d_in[1];
  const float* wordEmb = (const float*)d_in[2];
  const float* posEmb  = (const float*)d_in[3];
  const float* typeEmb = (const float*)d_in[4];
  const float* embLnG  = (const float*)d_in[5];
  const float* embLnB  = (const float*)d_in[6];
  auto LP = [&](int l, int j) { return (const float*)d_in[7 + l * 16 + j]; };

  char* base = (char*)d_ws;
  size_t off = 0;
  auto allocF = [&](size_t n) { float* p = (float*)(base + off); off += n * sizeof(float); return p; };
  auto allocH = [&](size_t n) { _Float16* p = (_Float16*)(base + off); off += n * sizeof(_Float16); return p; };

  float* TBL  = allocF((size_t)BB * VV * ATTR_N);    // vocab scatter table
  float* AO   = allocF((size_t)BB * ATTR_N * HD);    // attr_outs
  float* POOL = allocF((size_t)BB * ATTR_N * HD);    // pooled CLS
  float* EB   = allocF((size_t)NTOK);                // e scores
  float* WE   = allocF((size_t)NTOK * HD);           // gathered / pre-LN embeds (f32)
  float* HHb  = allocF((size_t)NTOK * HD);           // hh / bufO (f32)
  float* HB   = allocF((size_t)NTOK * HD);           // h (f32)
  float* QB   = allocF((size_t)NTOK * HD);
  float* KB   = allocF((size_t)NTOK * HD);
  float* VB   = allocF((size_t)NTOK * HD);
  _Float16* WE16  = allocH((size_t)NTOK * HD);       // f16 GEMM operands
  _Float16* HB16  = allocH((size_t)NTOK * HD);
  _Float16* CTX16 = allocH((size_t)NTOK * HD);
  _Float16* FB16  = allocH((size_t)NTOK * FFD);
  // f16 weights: per layer wq,wk,wv,wo (768x768), w1 (768x3072), w2 (3072x768); init w (768x768)
  _Float16* WQ16[NLAYER]; _Float16* WK16[NLAYER]; _Float16* WV16[NLAYER]; _Float16* WO16[NLAYER];
  _Float16* W116[NLAYER]; _Float16* W216[NLAYER];
  _Float16* WI16[ATTR_N];
  for (int l = 0; l < NLAYER; ++l) {
    WQ16[l] = allocH((size_t)HD * HD);  WK16[l] = allocH((size_t)HD * HD);
    WV16[l] = allocH((size_t)HD * HD);  WO16[l] = allocH((size_t)HD * HD);
    W116[l] = allocH((size_t)HD * FFD); W216[l] = allocH((size_t)FFD * HD);
  }
  for (int a = 0; a < ATTR_N; ++a) WI16[a] = allocH((size_t)HD * HD);

  auto cvt = [&](const float* src, _Float16* dst, int n) {
    cvt_f16_kernel<<<(n + 255) / 256, 256, 0, stream>>>(src, dst, n);
  };
  auto gemm = [&](const _Float16* A, const _Float16* W, const float* bias,
                  float* C, _Float16* C16, int M, int N, int K, int act) {
    dim3 grid(N / 128, M / 128);
    if (act) gemm_wmma_kernel<1><<<grid, 256, 0, stream>>>(A, W, bias, C, C16, M, N, K);
    else     gemm_wmma_kernel<0><<<grid, 256, 0, stream>>>(A, W, bias, C, C16, M, N, K);
  };

  // ---- weight conversion (once per launch) ----
  for (int l = 0; l < NLAYER; ++l) {
    cvt(LP(l, 0),  WQ16[l], HD * HD);
    cvt(LP(l, 2),  WK16[l], HD * HD);
    cvt(LP(l, 4),  WV16[l], HD * HD);
    cvt(LP(l, 6),  WO16[l], HD * HD);
    cvt(LP(l, 10), W116[l], HD * FFD);
    cvt(LP(l, 12), W216[l], FFD * HD);
  }
  for (int a = 0; a < ATTR_N; ++a)
    cvt((const float*)d_in[71 + a * 3 + 0], WI16[a], HD * HD);

  const int ntbl = BB * VV * ATTR_N;
  zero_kernel<<<(ntbl + 255) / 256, 256, 0, stream>>>(TBL, ntbl);

  // ---- stage 1: per-attribute init attention + vocab scatter ----
  for (int a = 0; a < ATTR_N; ++a) {
    const float* ib = (const float*)d_in[71 + a * 3 + 1];
    const float* ia = (const float*)d_in[71 + a * 3 + 2];
    const float* ca = (const float*)d_in[86 + a];
    gather_we_kernel<<<(NTOK * HD + 255) / 256, 256, 0, stream>>>(xs, wordEmb, WE, WE16, a);
    gemm(WE16, WI16[a], ib, HHb, nullptr, NTOK, HD, HD, 0);               // hh = we@w+b
    rowdot_leaky_kernel<<<NTOK / 8, 256, 0, stream>>>(HHb, ia, EB);       // e
    attr_pool_kernel<<<BB, 128, 0, stream>>>(EB, HHb, AO, a);             // attr_emb
    e2_scatter_kernel<<<NTOK / 8, 256, 0, stream>>>(WE, AO, ca, xs, TBL, a);
  }

  // ---- stage 2+3: per-attribute BERT encoder ----
  for (int a = 0; a < ATTR_N; ++a) {
    build_input_kernel<<<NTOK, 256, 0, stream>>>(xs, wordEmb, posEmb, typeEmb,
                                                 TBL, masks, AO, WE, a);
    ln_kernel<<<NTOK / 8, 256, 0, stream>>>(HB, HB16, WE, nullptr, embLnG, embLnB);
    for (int l = 0; l < NLAYER; ++l) {
      gemm(HB16, WQ16[l], LP(l, 1), QB, nullptr, NTOK, HD, HD, 0);        // q
      gemm(HB16, WK16[l], LP(l, 3), KB, nullptr, NTOK, HD, HD, 0);        // k
      gemm(HB16, WV16[l], LP(l, 5), VB, nullptr, NTOK, HD, HD, 0);        // v
      attention_kernel<<<BB * NHEAD, 128, 0, stream>>>(QB, KB, VB, CTX16);
      gemm(CTX16, WO16[l], LP(l, 7), HHb, nullptr, NTOK, HD, HD, 0);      // o proj
      ln_kernel<<<NTOK / 8, 256, 0, stream>>>(HB, HB16, HB, HHb, LP(l, 8), LP(l, 9));    // ln1
      gemm(HB16, W116[l], LP(l, 11), nullptr, FB16, NTOK, FFD, HD, 1);    // w1 + gelu (f16 out)
      gemm(FB16, W216[l], LP(l, 13), HHb, nullptr, NTOK, HD, FFD, 0);     // w2
      ln_kernel<<<NTOK / 8, 256, 0, stream>>>(HB, HB16, HB, HHb, LP(l, 14), LP(l, 15));  // ln2
    }
    copy_cls_kernel<<<BB, 256, 0, stream>>>(HB, POOL, a);
  }

  // ---- stage 4: output attention + logits ----
  final_kernel<<<BB, 256, 0, stream>>>(POOL, (const float*)d_in[91],
                                       (const float*)d_in[92], (const float*)d_in[93],
                                       (float*)d_out);
}
